// MahaClassifier_53584011985260
// MI455X (gfx1250) — matrix operand order repaired and checked by reading, not compiled, not run on hardware
//
#include <hip/hip_runtime.h>
#include <hip/hip_bf16.h>

#define P_ 8
#define T_ 200
#define F_ 512
#define C_ 20
#define Q_ 1024

typedef __attribute__((ext_vector_type(16))) _Float16 v16h;
typedef __attribute__((ext_vector_type(8)))  _Float16 v8h;
typedef __attribute__((ext_vector_type(8)))  float    v8f;
typedef __attribute__((ext_vector_type(4)))  unsigned int u32x4;
typedef __attribute__((ext_vector_type(8)))  int      i32x8;
typedef __attribute__((ext_vector_type(4)))  int      i32x4;

// workspace layout (float offsets)
#define WS_COUNTS   0
#define WS_MUTASK   32
#define WS_MUCLS    (WS_MUTASK + P_*F_)
#define WS_QUAD     (WS_MUCLS + (size_t)P_*C_*F_)
#define WS_COV      (WS_QUAD + (size_t)P_*Q_*C_)
#define WS_P16      (WS_COV + (size_t)P_*C_*F_*F_)   // f16 region, blocked layout

#if __has_builtin(__builtin_amdgcn_tensor_load_to_lds)
#define HAVE_TDM 1
#endif

__device__ __forceinline__ v16h cat16(v8h lo, v8h hi) {
    return __builtin_shufflevector(lo, hi, 0,1,2,3,4,5,6,7,8,9,10,11,12,13,14,15);
}

// ---------------------------------------------------------------------------
// Kernel 1: counts, per-class means, task means.  One block per phi.
// ---------------------------------------------------------------------------
__global__ __launch_bounds__(256) void k_means(
    const float* __restrict__ sf, const int* __restrict__ labels,
    float* __restrict__ ws, float* __restrict__ out)
{
    __shared__ float clsSum[C_*F_];
    __shared__ float taskSum[F_];
    __shared__ int   cntL[C_];
    __shared__ int   labL[T_];
    const int p = blockIdx.x;
    const int tid = threadIdx.x;

    for (int e = tid; e < C_*F_; e += 256) clsSum[e] = 0.f;
    for (int e = tid; e < F_;    e += 256) taskSum[e] = 0.f;
    if (tid < C_) cntL[tid] = 0;
    for (int e = tid; e < T_; e += 256) labL[e] = labels[e];
    __syncthreads();

    for (int t = 0; t < T_; ++t) {
        const int lab = labL[t];
        if (tid == 0) cntL[lab] += 1;
        for (int f = tid; f < F_; f += 256) {
            float v = sf[((size_t)p*T_ + t)*F_ + f];
            clsSum[lab*F_ + f] += v;   // column f owned by this thread: race-free
            taskSum[f]         += v;
        }
    }
    __syncthreads();

    for (int e = tid; e < C_*F_; e += 256) {
        int c = e >> 9, f = e & (F_-1);
        float n  = (float)cntL[c];
        float ns = n > 1.f ? n : 1.f;
        float mu = clsSum[e] / ns;
        ws[WS_MUCLS + ((size_t)p*C_ + c)*F_ + f] = mu;
        out[(size_t)Q_*C_ + ((size_t)p*C_ + c)*F_ + f] = mu;   // class_means output
    }
    for (int f = tid; f < F_; f += 256)
        ws[WS_MUTASK + (size_t)p*F_ + f] = taskSum[f] * (1.f / (float)T_);
    if (p == 0 && tid < C_) ws[WS_COUNTS + tid] = (float)cntL[tid];
}

// ---------------------------------------------------------------------------
// Kernel 2: cov_reg[p][c] = A^T A + I  via WMMA f16->f32.
// Stacked-row matrix A (K=416 incl. zero pad) staged K-slice at a time in
// LDS, stored [col][k] so each WMMA operand is two ds_load_b128.
// grid = P*C*8 (64-row strips), block = 512 (16 waves).
// ---------------------------------------------------------------------------
__global__ __launch_bounds__(512) void k_cov(
    const float* __restrict__ sf, const int* __restrict__ labels,
    float* __restrict__ ws)
{
    __shared__ __align__(16) _Float16 S[F_*32];   // [f][k]  32KB
    __shared__ float muC[F_];
    __shared__ float muT[F_];
    __shared__ int   labL[T_];

    const int tid = threadIdx.x;
    int b = blockIdx.x;
    const int strip = b & 7;  b >>= 3;
    const int c = b % C_;
    const int p = b / C_;
    const int r0 = strip * 64;

    for (int e = tid; e < T_; e += 512) labL[e] = labels[e];
    for (int f = tid; f < F_; f += 512) {
        muC[f] = ws[WS_MUCLS + ((size_t)p*C_ + c)*F_ + f];
        muT[f] = ws[WS_MUTASK + (size_t)p*F_ + f];
    }
    __syncthreads();

    const float cnt = ws[WS_COUNTS + c];
    const float ns  = fmaxf(cnt, 1.f);
    const float lam = ns / (ns + 1.f);
    const float dn  = fmaxf(cnt - 1.f, 1.f);
    const float sqs = sqrtf(lam / dn);
    const float sqt = sqrtf((1.f - lam) / (float)(T_ - 1));

    const int wid    = tid >> 5;
    const int lane   = tid & 31;
    const int laneHi = lane >> 4;
    const int nIdx   = lane & 15;
    const int rtile  = wid >> 2;   // 0..3  (16 G-rows each)
    const int colq   = wid & 3;    // 0..3  (8 col-tiles each)

    v8f acc[8] = {};

    for (int kb = 0; kb < 416; kb += 32) {
        // prefetch next K-slice rows (global_prefetch_b8) while we compute
        {
            int kn = kb + 32 + (tid >> 4);
            if (kn < 2*T_) {
                int tr = (kn < T_) ? kn : kn - T_;
                __builtin_prefetch(&sf[((size_t)p*T_ + tr)*F_ + ((tid & 15) << 5)], 0, 1);
            }
        }
        // build K-slice of the stacked matrix, transposed into [f][k]
        for (int e = tid; e < 32*F_; e += 512) {
            int kk = e >> 9;
            int f  = e & (F_-1);
            int k  = kb + kk;
            float val = 0.f;
            if (k < T_) {
                if (labL[k] == c)
                    val = sqs * (sf[((size_t)p*T_ + k)*F_ + f] - muC[f]);
            } else if (k < 2*T_) {
                int t = k - T_;
                val = sqt * (sf[((size_t)p*T_ + t)*F_ + f] - muT[f]);
            }
            S[f*32 + kk] = (_Float16)val;
        }
        __syncthreads();

        // A operand: rows r0+rtile*16+nIdx, K halves {8*laneHi+0..7, 16+8*laneHi+0..7}
        const _Float16* ab = &S[(r0 + rtile*16 + nIdx)*32];
        v16h aop = cat16(*(const v8h*)(ab + 8*laneHi),
                         *(const v8h*)(ab + 16 + 8*laneHi));

        #pragma unroll
        for (int tt = 0; tt < 8; ++tt) {
            // B operand: cols (colq*8+tt)*16+nIdx, K halves {8*laneHi+0..7, 16+8*laneHi+0..7}
            const _Float16* bb = &S[((colq*8 + tt)*16 + nIdx)*32];
            v16h bop = cat16(*(const v8h*)(bb + 8*laneHi), *(const v8h*)(bb + 16 + 8*laneHi));
            acc[tt] = __builtin_amdgcn_wmma_f32_16x16x32_f16(
                false, aop, false, bop, (short)0, acc[tt], false, false);
        }
        __syncthreads();
    }

    float* cov = ws + WS_COV + ((size_t)p*C_ + c)*F_*F_;
    #pragma unroll
    for (int tt = 0; tt < 8; ++tt) {
        int g = (colq*8 + tt)*16 + nIdx;
        #pragma unroll
        for (int r = 0; r < 8; ++r) {
            int row = r0 + rtile*16 + r + laneHi*8;
            cov[(size_t)row*F_ + g] = acc[tt][r] + (row == g ? 1.f : 0.f);
        }
    }
}

// ---------------------------------------------------------------------------
// Kernel 3: batched in-place Gauss-Jordan inverse (SPD, eigenvalues >= 1,
// no pivoting).  One 1024-thread workgroup per matrix; float4 row updates;
// the 160 x 1MB matrices stay resident in the 192MB L2.
// ---------------------------------------------------------------------------
__global__ __launch_bounds__(1024) void k_inv(float* __restrict__ ws)
{
    __shared__ float rowb[F_];
    __shared__ float colb[F_];
    float* A = ws + WS_COV + (size_t)blockIdx.x * F_ * F_;
    const int tid = threadIdx.x;

    for (int k = 0; k < F_; ++k) {
        if (tid < F_) rowb[tid]      = A[(size_t)k*F_ + tid];
        else          colb[tid - F_] = A[(size_t)(tid - F_)*F_ + k];
        __syncthreads();
        const float r = 1.f / rowb[k];
        for (int e4 = tid; e4 < (F_*F_)/4; e4 += 1024) {
            const int i  = e4 >> 7;
            const int j0 = (e4 & 127) << 2;
            float4 a = *(float4*)(A + (size_t)i*F_ + j0);
            if (i == k) {
                a.x = rowb[j0+0] * r;
                a.y = rowb[j0+1] * r;
                a.z = rowb[j0+2] * r;
                a.w = rowb[j0+3] * r;
                if (j0 <= k && k < j0 + 4) ((float*)&a)[k - j0] = r;
            } else {
                const float cr = colb[i] * r;
                a.x -= cr * rowb[j0+0];
                a.y -= cr * rowb[j0+1];
                a.z -= cr * rowb[j0+2];
                a.w -= cr * rowb[j0+3];
                if (j0 <= k && k < j0 + 4) ((float*)&a)[k - j0] = -cr;
            }
            *(float4*)(A + (size_t)i*F_ + j0) = a;
        }
        __syncthreads();
    }
}

// ---------------------------------------------------------------------------
// Kernel 3b: convert precision matrices to f16 in a blocked layout:
// slice s of (p,c) = rows k in [32s, 32s+32), stored [g][k], 32KB contiguous.
// grid = P*C*16, block = 256.
// ---------------------------------------------------------------------------
__global__ __launch_bounds__(256) void k_cvt(float* __restrict__ ws)
{
    const int b  = blockIdx.x;
    const int s  = b & 15;
    const int pc = b >> 4;
    const float*   src = ws + WS_COV + (size_t)pc*F_*F_ + (size_t)s*32*F_;
    _Float16*      dst = (_Float16*)(ws + WS_P16) + (size_t)b*32*F_;
    for (int e = threadIdx.x; e < 32*F_; e += 256) {
        int kk = e >> 9;
        int g  = e & (F_-1);
        dst[g*32 + kk] = (_Float16)src[(size_t)kk*F_ + g];
    }
}

// ---------------------------------------------------------------------------
// Kernel 4: quad[p][q][c] = d^T P d with d = mu_pc - q.
// K-loop outermost: the precision K-slice (pre-transposed f16, contiguous
// 32KB) is pulled into LDS by the Tensor Data Mover (tensor_load_to_lds +
// s_wait_tensorcnt), then E = D x P runs on WMMA with 4 accumulator tiles
// per wave; epilogue sums E .* D with shfl_xor half-wave reductions.
// grid = P*C*64 (16-query tiles), block = 256 (8 waves).
// ---------------------------------------------------------------------------
__global__ __launch_bounds__(256) void k_quad(
    const float* __restrict__ qf, float* __restrict__ ws)
{
    __shared__ __align__(16) _Float16 D16[16*F_];  // [q][f]  16KB
    __shared__ __align__(16) _Float16 Pl[F_*32];   // [g][k]  32KB
    __shared__ float qacc[16*8];                   // [q][wave]

    const int tid = threadIdx.x;
    int b = blockIdx.x;
    const int qt = b & 63;  b >>= 6;
    const int c = b % C_;
    const int p = b / C_;
    const int q0 = qt * 16;

    const float* muCls = ws + WS_MUCLS + ((size_t)p*C_ + c)*F_;
    for (int e = tid; e < 16*F_; e += 256) {
        int q = e >> 9, f = e & (F_-1);
        D16[e] = (_Float16)(muCls[f] - qf[((size_t)p*Q_ + q0 + q)*F_ + f]);
    }
    if (tid < 16*8) qacc[tid] = 0.f;
    __syncthreads();

    const int wid    = tid >> 5;
    const int lane   = tid & 31;
    const int laneHi = lane >> 4;
    const int nIdx   = lane & 15;

    const _Float16* p16 = (const _Float16*)(ws + WS_P16)
                        + (size_t)((p*C_ + c)*16) * 32 * F_;

    v8f acc[4] = {};

    for (int kb = 0; kb < F_; kb += 32) {
        const _Float16* src = p16 + (size_t)(kb >> 5) * 32 * F_;  // 32KB slice
#ifdef HAVE_TDM
        if (wid == 0) {
            // Tensor DMA descriptor: 1-D tile of 16384 x 2-byte elements
            const unsigned  lds = (unsigned)(size_t)(void*)&Pl[0];
            const unsigned long long ga = (unsigned long long)(size_t)(const void*)src;
            u32x4 g0 = { 1u,                                   // count=1
                         lds,                                  // lds_addr
                         (unsigned)ga,                         // global_addr lo
                         (unsigned)((ga >> 32) & 0x01FFFFFFu) | (2u << 30) }; // hi | type=2
            i32x8 g1 = { 0x10000,      // data_size=2B
                         0x40000000,   // tensor_dim0=16384 (lo16 in [31:16])
                         0x00010000,   // tensor_dim0 hi | tensor_dim1=1
                         0x40000000,   // tile_dim0=16384
                         1,            // tile_dim1=1
                         16384,        // tensor_dim0_stride lo
                         0x40000000,   // stride hi | tensor_dim1_stride lo
                         0 };
            i32x4 gz = { 0, 0, 0, 0 };
#if defined(__clang_major__) && __clang_major__ >= 23
            i32x8 gz8 = { 0, 0, 0, 0, 0, 0, 0, 0 };
            __builtin_amdgcn_tensor_load_to_lds(g0, g1, gz, gz, gz8, 0);
#else
            __builtin_amdgcn_tensor_load_to_lds(g0, g1, gz, gz, 0);
#endif
#if __has_builtin(__builtin_amdgcn_s_wait_tensorcnt)
            __builtin_amdgcn_s_wait_tensorcnt(0);
#endif
        }
#else
        for (int e = tid; e < (32*F_)/8; e += 256)
            ((uint4*)Pl)[e] = ((const uint4*)src)[e];
#endif
        __syncthreads();

        // A operand: query rows q = nIdx, K halves rel. to kb
        const _Float16* ab = &D16[nIdx*F_ + kb];
        v16h aop = cat16(*(const v8h*)(ab + 8*laneHi),
                         *(const v8h*)(ab + 16 + 8*laneHi));

        #pragma unroll
        for (int t = 0; t < 4; ++t) {
            const int ct = wid*4 + t;
            const _Float16* bb = &Pl[(ct*16 + nIdx)*32];
            v16h bop = cat16(*(const v8h*)(bb + 8*laneHi), *(const v8h*)(bb + 16 + 8*laneHi));
            acc[t] = __builtin_amdgcn_wmma_f32_16x16x32_f16(
                false, aop, false, bop, (short)0, acc[t], false, false);
        }
        __syncthreads();
    }

    // epilogue: quad_q += sum_g E[q][g] * d[q][g]
    #pragma unroll
    for (int t = 0; t < 4; ++t) {
        const int g = (wid*4 + t)*16 + nIdx;
        #pragma unroll
        for (int r = 0; r < 8; ++r) {
            int q = r + laneHi*8;
            float pr = acc[t][r] * (float)D16[q*F_ + g];
            pr += __shfl_xor(pr, 1, 32);
            pr += __shfl_xor(pr, 2, 32);
            pr += __shfl_xor(pr, 4, 32);
            pr += __shfl_xor(pr, 8, 32);
            if (nIdx == 0) qacc[q*8 + wid] += pr;   // unique (q,wave): race-free
        }
    }
    __syncthreads();
    if (tid < 16) {
        float s = 0.f;
        #pragma unroll
        for (int w = 0; w < 8; ++w) s += qacc[tid*8 + w];
        ws[WS_QUAD + ((size_t)p*Q_ + q0 + tid)*C_ + c] = s;
    }
}

// ---------------------------------------------------------------------------
// Kernel 5: mean over phi -> logits (deterministic serial sum over P=8)
// ---------------------------------------------------------------------------
__global__ __launch_bounds__(256) void k_logits(
    const float* __restrict__ ws, float* __restrict__ out)
{
    int idx = blockIdx.x * 256 + threadIdx.x;
    if (idx >= Q_*C_) return;
    int q = idx / C_, c = idx % C_;
    float s = 0.f;
    #pragma unroll
    for (int p = 0; p < P_; ++p)
        s += ws[WS_QUAD + ((size_t)p*Q_ + q)*C_ + c];
    out[idx] = -s * (1.f / (float)P_);
}

extern "C" void kernel_launch(void* const* d_in, const int* in_sizes, int n_in,
                              void* d_out, int out_size, void* d_ws, size_t ws_size,
                              hipStream_t stream) {
    (void)in_sizes; (void)n_in; (void)out_size; (void)ws_size;
    const float* sf     = (const float*)d_in[0];
    const int*   labels = (const int*)d_in[1];
    const float* qf     = (const float*)d_in[2];
    float* out = (float*)d_out;
    float* ws  = (float*)d_ws;

    k_means <<<P_,          256, 0, stream>>>(sf, labels, ws, out);
    k_cov   <<<P_*C_*8,     512, 0, stream>>>(sf, labels, ws);
    k_inv   <<<P_*C_,      1024, 0, stream>>>(ws);
    k_cvt   <<<P_*C_*16,    256, 0, stream>>>(ws);
    k_quad  <<<P_*C_*64,    256, 0, stream>>>(qf, ws);
    k_logits<<<(Q_*C_+255)/256, 256, 0, stream>>>(ws, out);
}
